// ExportEquiGraspSO3DeformableAttn_31044023615892
// MI455X (gfx1250) — compile-verified, matrix operand order checked by reading, not run on hardware
//
#include <hip/hip_runtime.h>

#define BS   8
#define NS   512
#define FD   512
#define NH   4
#define HID  128
#define NCP  8
#define RESO 64

#define SF_STRIDE 520   // bf16 elements per A-matrix row (512 + 8 pad); row = 1040 B (16B aligned)
#define KV_STRIDE 136   // bf16 elements per K/V row (128 + 8 pad)

// Packed-weight workspace layout (u32 units):
//   [mat(2)][nt(8)][kt(16)][lane(32)][vv(8)]
// Each u32 holds the bf16 pair (k0, k0+1) for that lane/vv slot of the WMMA
// B-fragment, so the main kernel loads a fragment as 8 consecutive u32.
#define WPK_PER_MAT (8 * 16 * 32 * 8)   // 32768 u32 = 128 KB per matrix

typedef __attribute__((ext_vector_type(16))) __bf16   v16bf;
typedef __attribute__((ext_vector_type(8)))  float    v8f;
typedef __attribute__((ext_vector_type(8)))  unsigned v8u;
typedef __attribute__((ext_vector_type(4)))  unsigned v4u;

static __device__ __forceinline__ __bf16 f2bf(float x) { return (__bf16)x; }
static __device__ __forceinline__ float  bf2f(__bf16 b) { return (float)b; }

// Bilinear sample of plane (BS, FD, RESO, RESO), uv in normalized [0,1] space.
static __device__ __forceinline__ float bilin(const float* __restrict__ plane,
                                              int b, int c, float u, float v) {
  float x = fminf(fmaxf(u, 0.f), 1.f) * (float)(RESO - 1);
  float y = fminf(fmaxf(v, 0.f), 1.f) * (float)(RESO - 1);
  float xf = floorf(x), yf = floorf(y);
  int x0 = (int)xf, y0 = (int)yf;
  int x1 = min(x0 + 1, RESO - 1), y1 = min(y0 + 1, RESO - 1);
  float wx = x - xf, wy = y - yf;
  const float* p = plane + (((size_t)b * FD + c) * RESO) * RESO;
  float f00 = p[y0 * RESO + x0];
  float f01 = p[y0 * RESO + x1];
  float f10 = p[y1 * RESO + x0];
  float f11 = p[y1 * RESO + x1];
  return f00 * (1.f - wx) * (1.f - wy) + f01 * wx * (1.f - wy)
       + f10 * (1.f - wx) * wy       + f11 * wx * wy;
}

// ---------------------------------------------------------------------------
// Pre-kernel: convert Wk/Wv (512x128 f32) to bf16 and pack into the WMMA
// B-fragment lane layout so the hot loop does zero conversion work.
// ---------------------------------------------------------------------------
__global__ __launch_bounds__(256)
void pack_weights_kernel(const float* __restrict__ Wk,
                         const float* __restrict__ Wv,
                         unsigned* __restrict__ wpk) {
  int idx  = blockIdx.x * 256 + threadIdx.x;   // 0 .. 2*WPK_PER_MAT-1
  int vv   = idx & 7;
  int lane = (idx >> 3) & 31;
  int kt   = (idx >> 8) & 15;
  int nt   = (idx >> 12) & 7;
  int mat  = idx >> 15;
  const float* W = mat ? Wv : Wk;
  int kh    = lane >> 4;
  int kbase = (vv < 4) ? (kh * 8 + 2 * vv) : (16 + kh * 8 + 2 * (vv - 4));
  int k0    = kt * 32 + kbase;
  int col   = nt * 16 + (lane & 15);
  union { unsigned u; __bf16 h[2]; } cv;
  cv.h[0] = f2bf(W[(size_t)k0 * HID + col]);
  cv.h[1] = f2bf(W[(size_t)(k0 + 1) * HID + col]);
  wpk[idx] = cv.u;
}

// ---------------------------------------------------------------------------
// Main kernel: one 256-thread workgroup (8 waves) per query.
// ---------------------------------------------------------------------------
__global__ __launch_bounds__(256)
void deform_attn_kernel(const float* __restrict__ query_pos,
                        const float* __restrict__ c_xy,
                        const float* __restrict__ c_xz,
                        const float* __restrict__ c_yz,
                        const float* __restrict__ ctrl,
                        const float* __restrict__ W_off,
                        const float* __restrict__ b_off,
                        const float* __restrict__ Wq,
                        const float* __restrict__ bq,
                        const float* __restrict__ bk,
                        const float* __restrict__ bv,
                        const float* __restrict__ Wo,
                        const float* __restrict__ bo,
                        const float* __restrict__ pe_q,
                        const float* __restrict__ pe_k,
                        const unsigned* __restrict__ Wk_pk,
                        const unsigned* __restrict__ Wv_pk,
                        float* __restrict__ out) {
  const int qidx = blockIdx.x;          // 0 .. BS*NS-1
  const int b    = qidx / NS;
  const int t    = threadIdx.x;

  __shared__ float  s_feat[FD];
  __shared__ __bf16 s_sfA[32 * SF_STRIDE];   // A-matrix (32 x 512) bf16
  __shared__ __bf16 s_K[32 * KV_STRIDE];     // K = A @ Wk (32 x 128)
  __shared__ __bf16 s_V[32 * KV_STRIDE];     // V = A @ Wv (32 x 128)
  __shared__ float  s_q[HID];
  __shared__ float  s_pt[32][3];             // sample point for A row r
  __shared__ float  s_off[NH * 3];
  __shared__ float  s_asp0[NCP][3];
  __shared__ float  s_sim[32];
  __shared__ float  s_attn[32];
  __shared__ float  s_o[NH * HID];

  const float* qp = query_pos + (size_t)qidx * 9;
  const float px = qp[0], py = qp[1], pz = qp[2];

  // ---------------- Phase A: feat[512] at the query position ----------------
  for (int c = t; c < FD; c += 256) {
    s_feat[c] = bilin(c_xy, b, c, px + 0.5f, py + 0.5f)
              + bilin(c_xz, b, c, px + 0.5f, pz + 0.5f)
              + bilin(c_yz, b, c, py + 0.5f, pz + 0.5f);
  }

  // rotation_6d_to_matrix + rotated control points (thread 0, independent of feat)
  if (t == 0) {
    float a1x = qp[3], a1y = qp[4], a1z = qp[5];
    float a2x = qp[6], a2y = qp[7], a2z = qp[8];
    float inv1 = 1.f / sqrtf(a1x * a1x + a1y * a1y + a1z * a1z);
    float b1x = a1x * inv1, b1y = a1y * inv1, b1z = a1z * inv1;
    float d = b1x * a2x + b1y * a2y + b1z * a2z;
    float b2x = a2x - d * b1x, b2y = a2y - d * b1y, b2z = a2z - d * b1z;
    float inv2 = 1.f / sqrtf(b2x * b2x + b2y * b2y + b2z * b2z);
    b2x *= inv2; b2y *= inv2; b2z *= inv2;
    float b3x = b1y * b2z - b1z * b2y;
    float b3y = b1z * b2x - b1x * b2z;
    float b3z = b1x * b2y - b1y * b2x;
    float R[3][3] = {{b1x, b1y, b1z}, {b2x, b2y, b2z}, {b3x, b3y, b3z}};
    for (int g = 0; g < NCP; ++g) {
      float cx = ctrl[g * 3 + 0], cy = ctrl[g * 3 + 1], cz = ctrl[g * 3 + 2];
      s_asp0[g][0] = px + R[0][0] * cx + R[0][1] * cy + R[0][2] * cz;
      s_asp0[g][1] = py + R[1][0] * cx + R[1][1] * cy + R[1][2] * cz;
      s_asp0[g][2] = pz + R[2][0] * cx + R[2][1] * cy + R[2][2] * cz;
    }
  }
  __syncthreads();

  // ---------------- Phase B: per-head offsets off = feat @ W_off + b_off ----
  if (t < NH * 3) {
    float acc = b_off[t];
    for (int f = 0; f < FD; ++f) acc += s_feat[f] * W_off[f * (NH * 3) + t];
    s_off[t] = acc;
  }
  __syncthreads();

  // A-matrix row r holds sample s=(r%8)*4 + r/8 (reference's (H,NCP)->(NCP,H) reshape)
  if (t < 32) {
    int s = (t & 7) * 4 + (t >> 3);
    int h = s >> 3, j = s & 7;
    for (int p = 0; p < 3; ++p) s_pt[t][p] = s_off[h * 3 + p] + s_asp0[j][p];
  }
  __syncthreads();

  // ---------------- Phase C: sample 32 points x 512 channels -> bf16 A ------
  for (int i = 0; i < 64; ++i) {
    int idx = i * 256 + t;
    int r = idx >> 9;        // / 512
    int f = idx & 511;
    float x = s_pt[r][0], y = s_pt[r][1], z = s_pt[r][2];
    float val = bilin(c_xy, b, f, x + 0.5f, y + 0.5f)
              + bilin(c_xz, b, f, x + 0.5f, z + 0.5f)
              + bilin(c_yz, b, f, y + 0.5f, z + 0.5f);
    s_sfA[r * SF_STRIDE + f] = f2bf(val);
  }

  // q = ((feat @ Wq + bq) + pe_q) / sqrt(HID)   (independent of sfA)
  if (t < HID) {
    float acc = bq[t] + pe_q[t];
    for (int f = 0; f < FD; ++f) acc += s_feat[f] * Wq[f * HID + t];
    s_q[t] = acc * 0.08838834764831845f;   // 1/sqrt(128)
  }
  __syncthreads();

  // ---------------- Phase D: K = A@Wk, V = A@Wv via bf16 WMMA ---------------
  // 8 waves x 2 (matrix,nt) column-groups; each B fragment feeds both M-tiles.
  {
    const int w    = t >> 5;
    const int lane = t & 31;
    const int m    = lane & 15;
    const int kh   = lane >> 4;          // K-half select per A/B layout
    const __bf16* arow0 = &s_sfA[m * SF_STRIDE];          // mt=0: A row m
    const __bf16* arow1 = &s_sfA[(16 + m) * SF_STRIDE];   // mt=1: A row 16+m
    for (int i = 0; i < 2; ++i) {
      const int pid  = w * 2 + i;        // 0..15
      const int isV  = pid >> 3;
      const int nt   = pid & 7;
      const int ncol = nt * 16 + m;
      const unsigned* Wpk = (isV ? Wv_pk : Wk_pk)
                          + ((size_t)(nt * 16) * 32 + lane) * 8;
      v8f acc0 = {}, acc1 = {};
#pragma unroll 2
      for (int kt = 0; kt < 16; ++kt) {
        union BF { v8u r; v16bf v; } bf;
        bf.r = *(const v8u*)(Wpk + (size_t)kt * 256);     // 2x b128, packed bf16
        union AF { struct { v4u lo, hi; } u; v16bf v; } a0, a1;
        const __bf16* p0 = arow0 + kt * 32 + kh * 8;
        const __bf16* p1 = arow1 + kt * 32 + kh * 8;
        a0.u.lo = *(const v4u*)p0;  a0.u.hi = *(const v4u*)(p0 + 16);
        a1.u.lo = *(const v4u*)p1;  a1.u.hi = *(const v4u*)(p1 + 16);
        acc0 = __builtin_amdgcn_wmma_f32_16x16x32_bf16(
            false, a0.v, false, bf.v, (short)0, acc0, false, false);
        acc1 = __builtin_amdgcn_wmma_f32_16x16x32_bf16(
            false, a1.v, false, bf.v, (short)0, acc1, false, false);
      }
      __bf16* Dmat = isV ? s_V : s_K;
#pragma unroll
      for (int vv = 0; vv < 8; ++vv) {
        int mrow = vv + kh * 8;          // D layout: VGPR v -> M = v (+8)
        Dmat[mrow * KV_STRIDE + ncol]        = f2bf(acc0[vv]);
        Dmat[(16 + mrow) * KV_STRIDE + ncol] = f2bf(acc1[vv]);
      }
    }
  }
  __syncthreads();

  // ---------------- Phase F: sim = q . (k + bk + pe_k), softmax over 8 ------
  if (t < 32) {
    int jj = t & 7;                     // row r == t == hh*8 + jj
    float acc = 0.f;
    for (int d = 0; d < HID; ++d) {
      float kv = bf2f(s_K[t * KV_STRIDE + d]) + bk[d] + pe_k[jj * HID + d];
      acc += s_q[d] * kv;
    }
    s_sim[t] = acc;
  }
  __syncthreads();
  if (t < NH) {
    float mx = s_sim[t * 8];
    for (int j = 1; j < NCP; ++j) mx = fmaxf(mx, s_sim[t * 8 + j]);
    float sum = 0.f, e[NCP];
    for (int j = 0; j < NCP; ++j) { e[j] = __expf(s_sim[t * 8 + j] - mx); sum += e[j]; }
    float inv = 1.f / sum;
    for (int j = 0; j < NCP; ++j) s_attn[t * 8 + j] = e[j] * inv;
  }
  __syncthreads();

  // ---------------- Phase G: out_h = attn @ (V + bv) ------------------------
  for (int idx = t; idx < NH * HID; idx += 256) {
    int hh = idx >> 7, d = idx & 127;
    float acc = 0.f;
    for (int j = 0; j < NCP; ++j) {
      float vv = bf2f(s_V[(hh * 8 + j) * KV_STRIDE + d]) + bv[d];
      acc += s_attn[hh * 8 + j] * vv;
    }
    s_o[hh * HID + d] = acc;
  }
  __syncthreads();

  // ---------------- Phase H: final = o @ Wo + bo + feat ---------------------
  __builtin_prefetch(Wo, 0, 1);
  for (int c = t; c < FD; c += 256) {
    float acc = bo[c] + s_feat[c];
    for (int i = 0; i < NH * HID; ++i) acc += s_o[i] * Wo[(size_t)i * FD + c];
    out[(size_t)qidx * FD + c] = acc;
  }
}

extern "C" void kernel_launch(void* const* d_in, const int* in_sizes, int n_in,
                              void* d_out, int out_size, void* d_ws, size_t ws_size,
                              hipStream_t stream) {
  (void)in_sizes; (void)n_in; (void)out_size; (void)ws_size;
  const float* Wk = (const float*)d_in[9];
  const float* Wv = (const float*)d_in[11];
  unsigned* wpk = (unsigned*)d_ws;                 // 2 * 128 KB used

  pack_weights_kernel<<<(2 * WPK_PER_MAT) / 256, 256, 0, stream>>>(Wk, Wv, wpk);

  deform_attn_kernel<<<BS * NS, 256, 0, stream>>>(
      (const float*)d_in[0],  (const float*)d_in[1],  (const float*)d_in[2],
      (const float*)d_in[3],  (const float*)d_in[4],  (const float*)d_in[5],
      (const float*)d_in[6],  (const float*)d_in[7],  (const float*)d_in[8],
      (const float*)d_in[10], (const float*)d_in[12], (const float*)d_in[13],
      (const float*)d_in[14], (const float*)d_in[15], (const float*)d_in[16],
      wpk, wpk + WPK_PER_MAT, (float*)d_out);
}
